// TransformerDecoderLayer_87101936763705
// MI455X (gfx1250) — compile-verified, matrix operand order checked by reading, not run on hardware
//
#include <hip/hip_runtime.h>
#include <cstdint>
#include <cstddef>

// ---------------- constants (match reference) ----------------
#define BB   8
#define SEQ  1024
#define EDIM 1024
#define NH   16
#define HD   64
#define FF   4096
#define MROWS (BB * SEQ)   // 8192

typedef __attribute__((ext_vector_type(16))) __bf16 v16bf;
typedef __attribute__((ext_vector_type(8)))  __bf16 v8bf;
typedef __attribute__((ext_vector_type(8)))  float  v8f;
typedef __attribute__((ext_vector_type(4)))  unsigned int u32x4;
typedef __attribute__((ext_vector_type(8)))  int          i32x8;
typedef __attribute__((ext_vector_type(4)))  int          i32x4;

union AFrag { v16bf v; v8bf h[2]; };

// -------- CDNA5 data-mover feature detection (device pass only) --------
#if defined(__HIP_DEVICE_COMPILE__) && __has_builtin(__builtin_amdgcn_global_load_async_to_lds_b128) && __has_builtin(__builtin_amdgcn_s_wait_asynccnt)
#define USE_ASYNC 1
#else
#define USE_ASYNC 0
#endif
#if defined(__HIP_DEVICE_COMPILE__) && __has_builtin(__builtin_amdgcn_tensor_load_to_lds) && __has_builtin(__builtin_amdgcn_s_wait_tensorcnt)
#define USE_TDM 1
#else
#define USE_TDM 0
#endif

#if USE_ASYNC
#define GLOBAL_AS __attribute__((address_space(1)))
#define LDS_AS    __attribute__((address_space(3)))
static __device__ __forceinline__ void async_b128(const void* g, void* l) {
    // builtin signature (from hipcc diagnostic): (int4 addrspace(1)*, int4 addrspace(3)*, imm, imm)
    __builtin_amdgcn_global_load_async_to_lds_b128(
        (GLOBAL_AS i32x4*)g, (LDS_AS i32x4*)l, 0, 0);
}
#endif

#if USE_TDM
static __device__ __forceinline__ unsigned lds_off(const void* p) {
    return (unsigned)(size_t)(__attribute__((address_space(3))) const void*)p;
}
// 2D tensor -> LDS tile via Tensor Data Mover.  data_size=2B.
// pad_interval=16 DWORDs (code 3), pad_amount=4 DWORDs (code 3) => LDS row
// stride = tile0 + 8 halves, matching the padded shared tiles below.
static __device__ __forceinline__ void tdm_load_2d(unsigned lds_byte, const void* gp,
                                                   int dim0, int dim1, int tile0, int tile1) {
    unsigned long long ga = (unsigned long long)(size_t)gp;
    u32x4 g0;
    g0[0] = 1u;                                          // count=1 (valid user D#)
    g0[1] = lds_byte;                                    // lds_addr (bytes)
    g0[2] = (unsigned)(ga & 0xFFFFFFFFu);                // global_addr[31:0]
    g0[3] = (unsigned)((ga >> 32) & 0x1FFFFFFu) | (2u << 30);  // addr[56:32] | type=2
    i32x8 g1;
    g1[0] = (1 << 16) | (1 << 20) | (3 << 22) | (3 << 25);     // 2B elems, pad on
    g1[1] = (int)((unsigned)(dim0 & 0xFFFF) << 16);            // tensor_dim0[15:0]
    g1[2] = (int)(((unsigned)dim0 >> 16) | ((unsigned)(dim1 & 0xFFFF) << 16));
    g1[3] = (int)(((unsigned)dim1 >> 16) | ((unsigned)tile0 << 16));
    g1[4] = tile1 & 0xFFFF;                              // tile_dim1 (tile_dim2=0)
    g1[5] = dim0;                                        // tensor_dim0_stride[31:0]
    g1[6] = 0;
    g1[7] = 0;
    i32x4 gz = {0, 0, 0, 0};
#if defined(__clang_major__) && __clang_major__ >= 23
    i32x8 gz8 = {0, 0, 0, 0, 0, 0, 0, 0};
    __builtin_amdgcn_tensor_load_to_lds(g0, g1, gz, gz, gz8, 0);
#else
    __builtin_amdgcn_tensor_load_to_lds(g0, g1, gz, gz, 0);
#endif
}
#endif

static __device__ __forceinline__ __bf16 f2bf(float f) {
    unsigned u = __builtin_bit_cast(unsigned, f);
    u += 0x7FFFu + ((u >> 16) & 1u);           // round-to-nearest-even
    unsigned short s = (unsigned short)(u >> 16);
    return __builtin_bit_cast(__bf16, s);
}

// ---------------- fp32 -> bf16 conversion kernels ----------------
__global__ __launch_bounds__(256) void k_conv_plain(const float* __restrict__ x,
                                                    __bf16* __restrict__ o, size_t n) {
    for (size_t i = (size_t)blockIdx.x * 256 + threadIdx.x; i < n;
         i += (size_t)gridDim.x * 256)
        o[i] = f2bf(x[i]);
}

// W [K,N] fp32 -> out [N,K] bf16 (transposed for contiguous-K B fragments)
__global__ __launch_bounds__(256) void k_conv_w_t(const float* __restrict__ W,
                                                  __bf16* __restrict__ o, int K, int N) {
    size_t total = (size_t)K * N;
    for (size_t i = (size_t)blockIdx.x * 256 + threadIdx.x; i < total;
         i += (size_t)gridDim.x * 256) {
        size_t n = i / (size_t)K;
        size_t k = i - n * (size_t)K;
        o[i] = f2bf(W[k * (size_t)N + n]);
    }
}

// Wh [H,E,D] fp32 -> out [N=E, K=E] bf16 where column n = h*D+d
__global__ __launch_bounds__(256) void k_conv_wh_t(const float* __restrict__ Wh,
                                                   __bf16* __restrict__ o) {
    size_t total = (size_t)EDIM * EDIM;
    for (size_t i = (size_t)blockIdx.x * 256 + threadIdx.x; i < total;
         i += (size_t)gridDim.x * 256) {
        size_t n = i >> 10;            // / EDIM
        size_t e = i & 1023;           // % EDIM
        size_t h = n >> 6;             // / HD
        size_t d = n & 63;             // % HD
        o[i] = f2bf(Wh[(h * EDIM + e) * HD + d]);
    }
}

// ---------------- layernorm (fp32 in -> bf16 out) ----------------
__global__ __launch_bounds__(256) void k_layernorm(const float* __restrict__ x,
                                                   const float* __restrict__ g,
                                                   const float* __restrict__ bta,
                                                   __bf16* __restrict__ out) {
    const int row = blockIdx.x;
    const float* xr = x + (size_t)row * EDIM;
    float s = 0.f, s2 = 0.f;
    for (int i = threadIdx.x; i < EDIM; i += 256) {
        float v = xr[i]; s += v; s2 += v * v;
    }
    #pragma unroll
    for (int off = 16; off > 0; off >>= 1) {
        s  += __shfl_xor(s,  off, 32);
        s2 += __shfl_xor(s2, off, 32);
    }
    __shared__ float red[18];
    int wv = threadIdx.x >> 5, ln = threadIdx.x & 31;
    if (ln == 0) { red[wv] = s; red[8 + wv] = s2; }
    __syncthreads();
    if (threadIdx.x == 0) {
        float a = 0.f, b2 = 0.f;
        #pragma unroll
        for (int w = 0; w < 8; w++) { a += red[w]; b2 += red[8 + w]; }
        red[16] = a; red[17] = b2;
    }
    __syncthreads();
    float mean = red[16] * (1.0f / EDIM);
    float var  = red[17] * (1.0f / EDIM) - mean * mean;
    float inv  = rsqrtf(var + 1e-5f);
    for (int i = threadIdx.x; i < EDIM; i += 256)
        out[(size_t)row * EDIM + i] = f2bf((xr[i] - mean) * inv * g[i] + bta[i]);
}

// ---------------- tiled bf16 WMMA GEMM (double-buffered TDM pipeline) -------
// C = A[M,K] * Bt[N,K]^T + bias ; EPI: 0=bias->bf16, 1=bias+GELU->bf16, 2=bias+residual->f32
#define BM 128
#define BN 128
#define BK 32
template <int EPI>
__global__ __launch_bounds__(256) void k_gemm_bf16(const __bf16* __restrict__ A,
                                                   const __bf16* __restrict__ Bt,
                                                   const float* __restrict__ bias,
                                                   const float* __restrict__ res,
                                                   __bf16* __restrict__ Cbf,
                                                   float* __restrict__ Cf,
                                                   int M, int N, int K) {
    __shared__ __bf16 sA[2][BM][BK + 8];   // 2 x 10KB
    __shared__ __bf16 sB[2][BN][BK + 8];   // 2 x 10KB
    const int tid  = threadIdx.x;
    const int wv   = tid >> 5;
    const int lane = tid & 31;
    const int hf   = lane >> 4;
    const int l16  = lane & 15;
    const int wm   = wv & 3;     // 4 waves along M -> 32 rows each
    const int wn   = wv >> 2;    // 2 waves along N -> 64 cols each
    const int bm0  = blockIdx.y * BM;
    const int bn0  = blockIdx.x * BN;

    const v8f z8 = {0.f, 0.f, 0.f, 0.f, 0.f, 0.f, 0.f, 0.f};
    v8f acc[2][4];
    #pragma unroll
    for (int i = 0; i < 2; i++)
        #pragma unroll
        for (int j = 0; j < 4; j++) acc[i][j] = z8;

    const int lrow = tid >> 1;
    const int lcol = (tid & 1) * 16;
    (void)lrow; (void)lcol;

    // stage tile pair (A,B) at column k0 into LDS buffer w
    auto stage = [&](int w, int k0) {
#if USE_TDM
        if (tid < 32) {   // one wave drives the Tensor Data Mover
            tdm_load_2d(lds_off(&sA[w][0][0]), A  + (size_t)bm0 * K + (size_t)k0, K, M, BK, BM);
            tdm_load_2d(lds_off(&sB[w][0][0]), Bt + (size_t)bn0 * K + (size_t)k0, K, N, BK, BN);
        }
#elif USE_ASYNC
        const __bf16* sa = A  + (size_t)(bm0 + lrow) * K + k0 + lcol;
        const __bf16* sb = Bt + (size_t)(bn0 + lrow) * K + k0 + lcol;
        async_b128(sa,     &sA[w][lrow][lcol]);
        async_b128(sa + 8, &sA[w][lrow][lcol + 8]);
        async_b128(sb,     &sB[w][lrow][lcol]);
        async_b128(sb + 8, &sB[w][lrow][lcol + 8]);
#else
        const __bf16* sa = A  + (size_t)(bm0 + lrow) * K + k0 + lcol;
        const __bf16* sb = Bt + (size_t)(bn0 + lrow) * K + k0 + lcol;
        *(v8bf*)&sA[w][lrow][lcol]     = *(const v8bf*)(sa);
        *(v8bf*)&sA[w][lrow][lcol + 8] = *(const v8bf*)(sa + 8);
        *(v8bf*)&sB[w][lrow][lcol]     = *(const v8bf*)(sb);
        *(v8bf*)&sB[w][lrow][lcol + 8] = *(const v8bf*)(sb + 8);
        __builtin_prefetch(sa + BK, 0, 1);
        __builtin_prefetch(sb + BK, 0, 1);
#endif
    };

    const int NT = K / BK;
    stage(0, 0);                                     // prologue: fill buffer 0
    for (int kt = 0; kt < NT; kt++) {
        const int cur = kt & 1;
        // ---- software pipeline: issue next tile's DMA, then drain current ----
        if (kt + 1 < NT) {
            stage(cur ^ 1, (kt + 1) * BK);
#if USE_TDM
            __builtin_amdgcn_s_wait_tensorcnt(2);    // in-order: current pair landed
#elif USE_ASYNC
            __builtin_amdgcn_s_wait_asynccnt(4);     // 4 outstanding = next tile only
#endif
        } else {
#if USE_TDM
            __builtin_amdgcn_s_wait_tensorcnt(0);
#elif USE_ASYNC
            __builtin_amdgcn_s_wait_asynccnt(0);
#endif
        }
        __syncthreads();

        const int kk = hf * 8;               // CDNA5 16-bit A/B fragment K base
        AFrag afr[2], bfr[4];
        #pragma unroll
        for (int i = 0; i < 2; i++) {
            int r = wm * 32 + i * 16 + l16;
            afr[i].h[0] = *(const v8bf*)&sA[cur][r][kk];
            afr[i].h[1] = *(const v8bf*)&sA[cur][r][kk + 16];
        }
        #pragma unroll
        for (int j = 0; j < 4; j++) {
            int c = wn * 64 + j * 16 + l16;
            bfr[j].h[0] = *(const v8bf*)&sB[cur][c][kk];
            bfr[j].h[1] = *(const v8bf*)&sB[cur][c][kk + 16];
        }
        #pragma unroll
        for (int i = 0; i < 2; i++)
            #pragma unroll
            for (int j = 0; j < 4; j++)
                acc[i][j] = __builtin_amdgcn_wmma_f32_16x16x32_bf16(
                    false, afr[i].v, false, bfr[j].v, (short)0, acc[i][j], false, false);
        __syncthreads();   // all reads of buffer `cur` done before it is restaged
    }

    // epilogue: C layout -> lane = N, vgpr v = M (+8 for upper half-wave)
    #pragma unroll
    for (int i = 0; i < 2; i++) {
        #pragma unroll
        for (int j = 0; j < 4; j++) {
            int col = bn0 + wn * 64 + j * 16 + l16;
            float bv = bias[col];
            #pragma unroll
            for (int v = 0; v < 8; v++) {
                int row = bm0 + wm * 32 + i * 16 + v + hf * 8;
                float x = acc[i][j][v] + bv;
                if (EPI == 1) x = 0.5f * x * (1.0f + erff(x * 0.70710678118654752f));
                if (EPI == 2)
                    Cf[(size_t)row * N + col] = x + res[(size_t)row * N + col];
                else
                    Cbf[(size_t)row * N + col] = f2bf(x);
            }
        }
    }
}

// ---------------- flash attention (bf16 WMMA, online softmax) ----------------
// Q/K/V strided [b, s, h*HD + d] with row stride EDIM; O same layout.
template <bool CAUSAL>
__global__ __launch_bounds__(128) void k_flash(const __bf16* __restrict__ Qh,
                                               const __bf16* __restrict__ Kh,
                                               const __bf16* __restrict__ Vh,
                                               __bf16* __restrict__ O,
                                               int Sq, int Sk) {
    __shared__ __bf16 sK[64][72];
    __shared__ __bf16 sVt[64][72];
    __shared__ __bf16 sP[4][16][72];

    const int tid  = threadIdx.x;
    const int wv   = tid >> 5;
    const int lane = tid & 31;
    const int hf   = lane >> 4;
    const int l16  = lane & 15;
    const int qb   = blockIdx.x;
    const int h    = blockIdx.y;
    const int b    = blockIdx.z;
    const size_t ho = (size_t)h * HD;

    const __bf16* Qb = Qh + ((size_t)b * Sq) * EDIM + ho;
    const __bf16* Kb = Kh + ((size_t)b * Sk) * EDIM + ho;
    const __bf16* Vb = Vh + ((size_t)b * Sk) * EDIM + ho;

    // Q fragments live in registers for the whole kernel (2 k-steps over D=64)
    AFrag qf[2];
    {
        const __bf16* qp = Qb + (size_t)(qb * 64 + wv * 16 + l16) * EDIM;
        #pragma unroll
        for (int ks = 0; ks < 2; ks++) {
            int kk = ks * 32 + hf * 8;
            qf[ks].h[0] = *(const v8bf*)(qp + kk);
            qf[ks].h[1] = *(const v8bf*)(qp + kk + 16);
        }
    }

    const v8f z8 = {0.f, 0.f, 0.f, 0.f, 0.f, 0.f, 0.f, 0.f};
    v8f o_acc[4];
    float m_i[8], l_i[8];
    #pragma unroll
    for (int j = 0; j < 4; j++) o_acc[j] = z8;
    #pragma unroll
    for (int v = 0; v < 8; v++) { m_i[v] = -3.0e38f; l_i[v] = 0.f; }

    const int kbmax = CAUSAL ? qb : (Sk / 64 - 1);
    for (int kb = 0; kb <= kbmax; kb++) {
        // cooperative stage: K tile row-major (async DMA), V tile transposed [d][key]
        {
            int row = tid >> 1;
            int c0  = (tid & 1) * 32;
            const __bf16* kp = Kb + (size_t)(kb * 64 + row) * EDIM + c0;
            const __bf16* vp = Vb + (size_t)(kb * 64 + row) * EDIM + c0;
#if USE_ASYNC
            #pragma unroll
            for (int c = 0; c < 32; c += 8)
                async_b128(kp + c, &sK[row][c0 + c]);
#else
            #pragma unroll
            for (int c = 0; c < 32; c += 8)
                *(v8bf*)&sK[row][c0 + c] = *(const v8bf*)(kp + c);
#endif
            #pragma unroll
            for (int c = 0; c < 32; c++)
                sVt[c0 + c][row] = vp[c];
        }
#if USE_ASYNC
        __builtin_amdgcn_s_wait_asynccnt(0);
#endif
        __syncthreads();

        // S = Q K^T
        v8f sc[4];
        #pragma unroll
        for (int j = 0; j < 4; j++) sc[j] = z8;
        #pragma unroll
        for (int ks = 0; ks < 2; ks++) {
            int kk = ks * 32 + hf * 8;
            #pragma unroll
            for (int j = 0; j < 4; j++) {
                AFrag bf;
                int key = j * 16 + l16;
                bf.h[0] = *(const v8bf*)&sK[key][kk];
                bf.h[1] = *(const v8bf*)&sK[key][kk + 16];
                sc[j] = __builtin_amdgcn_wmma_f32_16x16x32_bf16(
                    false, qf[ks].v, false, bf.v, (short)0, sc[j], false, false);
            }
        }

        // online softmax (row = v + hf*8, its 16 keys live across lanes of the half-wave)
        #pragma unroll
        for (int v = 0; v < 8; v++) {
            int qrow = qb * 64 + wv * 16 + v + hf * 8;
            float mc = -3.0e38f;
            #pragma unroll
            for (int j = 0; j < 4; j++) {
                float s = sc[j][v] * 0.125f;      // 1/sqrt(64)
                if (CAUSAL) {
                    int key = kb * 64 + j * 16 + l16;
                    if (key > qrow) s = -3.0e38f;
                }
                sc[j][v] = s;
                mc = fmaxf(mc, s);
            }
            #pragma unroll
            for (int off = 1; off < 16; off <<= 1)
                mc = fmaxf(mc, __shfl_xor(mc, off, 16));
            float mnew  = fmaxf(m_i[v], mc);
            float alpha = __expf(m_i[v] - mnew);
            float rs = 0.f;
            #pragma unroll
            for (int j = 0; j < 4; j++) {
                float p = __expf(sc[j][v] - mnew);
                rs += p;
                sP[wv][v + hf * 8][j * 16 + l16] = f2bf(p);
            }
            #pragma unroll
            for (int off = 1; off < 16; off <<= 1)
                rs += __shfl_xor(rs, off, 16);
            l_i[v] = l_i[v] * alpha + rs;
            m_i[v] = mnew;
            #pragma unroll
            for (int j = 0; j < 4; j++) o_acc[j][v] *= alpha;
        }
        __syncthreads();

        // O += P V   (P re-read from LDS in A-fragment layout, V^T as B)
        #pragma unroll
        for (int ks = 0; ks < 2; ks++) {
            int kk = ks * 32 + hf * 8;
            AFrag pf;
            pf.h[0] = *(const v8bf*)&sP[wv][l16][kk];
            pf.h[1] = *(const v8bf*)&sP[wv][l16][kk + 16];
            #pragma unroll
            for (int j = 0; j < 4; j++) {
                AFrag bf;
                int d = j * 16 + l16;
                bf.h[0] = *(const v8bf*)&sVt[d][kk];
                bf.h[1] = *(const v8bf*)&sVt[d][kk + 16];
                o_acc[j] = __builtin_amdgcn_wmma_f32_16x16x32_bf16(
                    false, pf.v, false, bf.v, (short)0, o_acc[j], false, false);
            }
        }
        __syncthreads();
    }

    // write O (normalized), heads concatenated in-place
    {
        int q0 = qb * 64 + wv * 16;
        #pragma unroll
        for (int j = 0; j < 4; j++) {
            int d = j * 16 + l16;
            #pragma unroll
            for (int v = 0; v < 8; v++) {
                int qrow = q0 + v + hf * 8;
                O[((size_t)b * Sq + qrow) * EDIM + ho + d] = f2bf(o_acc[j][v] / l_i[v]);
            }
        }
    }
}

// ---------------- host orchestration ----------------
extern "C" void kernel_launch(void* const* d_in, const int* in_sizes, int n_in,
                              void* d_out, int out_size, void* d_ws, size_t ws_size,
                              hipStream_t stream) {
    (void)in_sizes; (void)n_in; (void)out_size; (void)ws_size;

    const float* src = (const float*)d_in[0];
    const float* tgt = (const float*)d_in[1];
    // d_in[2], d_in[3]: masks (causal / all-false) — handled structurally.
    const float *Wq1=(const float*)d_in[4],  *bq1=(const float*)d_in[5];
    const float *Wk1=(const float*)d_in[6],  *bk1=(const float*)d_in[7];
    const float *Wv1=(const float*)d_in[8],  *bv1=(const float*)d_in[9];
    const float *Whq1=(const float*)d_in[10],*bhq1=(const float*)d_in[11];
    const float *Whk1=(const float*)d_in[12],*bhk1=(const float*)d_in[13];
    const float *Whv1=(const float*)d_in[14],*bhv1=(const float*)d_in[15];
    const float *Wo1=(const float*)d_in[16], *bo1=(const float*)d_in[17];
    const float *Wq2=(const float*)d_in[18], *bq2=(const float*)d_in[19];
    const float *Wk2=(const float*)d_in[20], *bk2=(const float*)d_in[21];
    const float *Wv2=(const float*)d_in[22], *bv2=(const float*)d_in[23];
    const float *Whq2=(const float*)d_in[24],*bhq2=(const float*)d_in[25];
    const float *Whk2=(const float*)d_in[26],*bhk2=(const float*)d_in[27];
    const float *Whv2=(const float*)d_in[28],*bhv2=(const float*)d_in[29];
    const float *Wo2=(const float*)d_in[30], *bo2=(const float*)d_in[31];
    const float *W1=(const float*)d_in[32],  *b1=(const float*)d_in[33];
    const float *W2=(const float*)d_in[34],  *b2=(const float*)d_in[35];
    const float *g1=(const float*)d_in[36],  *be1=(const float*)d_in[37];
    const float *g2=(const float*)d_in[38],  *be2=(const float*)d_in[39];
    const float *g3=(const float*)d_in[40],  *be3=(const float*)d_in[41];

    char* ws = (char*)d_ws;
    size_t off = 0;
    auto alloc = [&](size_t bytes) -> void* {
        void* p = ws + off;
        off += (bytes + 255) & ~(size_t)255;
        return p;
    };
    const size_t EE2 = (size_t)EDIM * EDIM * 2;
    const size_t ME2 = (size_t)MROWS * EDIM * 2;
    const size_t MF2 = (size_t)MROWS * FF * 2;
    const size_t EF2 = (size_t)EDIM * FF * 2;
    const size_t ME4 = (size_t)MROWS * EDIM * 4;

    __bf16 *wq1=(__bf16*)alloc(EE2), *wk1=(__bf16*)alloc(EE2), *wv1=(__bf16*)alloc(EE2);
    __bf16 *whq1=(__bf16*)alloc(EE2),*whk1=(__bf16*)alloc(EE2),*whv1=(__bf16*)alloc(EE2);
    __bf16 *wo1=(__bf16*)alloc(EE2);
    __bf16 *wq2=(__bf16*)alloc(EE2), *wk2=(__bf16*)alloc(EE2), *wv2=(__bf16*)alloc(EE2);
    __bf16 *whq2=(__bf16*)alloc(EE2),*whk2=(__bf16*)alloc(EE2),*whv2=(__bf16*)alloc(EE2);
    __bf16 *wo2=(__bf16*)alloc(EE2);
    __bf16 *w1t=(__bf16*)alloc(EF2), *w2t=(__bf16*)alloc(EF2);
    __bf16 *srcbf=(__bf16*)alloc(ME2);
    __bf16 *xn=(__bf16*)alloc(ME2);
    __bf16 *qb=(__bf16*)alloc(ME2), *kb=(__bf16*)alloc(ME2), *vb=(__bf16*)alloc(ME2);
    __bf16 *qhb=(__bf16*)alloc(ME2),*khb=(__bf16*)alloc(ME2),*vhb=(__bf16*)alloc(ME2);
    __bf16 *ob=(__bf16*)alloc(ME2);
    __bf16 *hb=(__bf16*)alloc(MF2);
    float  *r1=(float*)alloc(ME4), *r2=(float*)alloc(ME4);
    float  *out = (float*)d_out;

    const dim3 cg(2048);
    k_conv_w_t <<<cg, 256, 0, stream>>>(Wq1, wq1, EDIM, EDIM);
    k_conv_w_t <<<cg, 256, 0, stream>>>(Wk1, wk1, EDIM, EDIM);
    k_conv_w_t <<<cg, 256, 0, stream>>>(Wv1, wv1, EDIM, EDIM);
    k_conv_wh_t<<<cg, 256, 0, stream>>>(Whq1, whq1);
    k_conv_wh_t<<<cg, 256, 0, stream>>>(Whk1, whk1);
    k_conv_wh_t<<<cg, 256, 0, stream>>>(Whv1, whv1);
    k_conv_w_t <<<cg, 256, 0, stream>>>(Wo1, wo1, EDIM, EDIM);
    k_conv_w_t <<<cg, 256, 0, stream>>>(Wq2, wq2, EDIM, EDIM);
    k_conv_w_t <<<cg, 256, 0, stream>>>(Wk2, wk2, EDIM, EDIM);
    k_conv_w_t <<<cg, 256, 0, stream>>>(Wv2, wv2, EDIM, EDIM);
    k_conv_wh_t<<<cg, 256, 0, stream>>>(Whq2, whq2);
    k_conv_wh_t<<<cg, 256, 0, stream>>>(Whk2, whk2);
    k_conv_wh_t<<<cg, 256, 0, stream>>>(Whv2, whv2);
    k_conv_w_t <<<cg, 256, 0, stream>>>(Wo2, wo2, EDIM, EDIM);
    k_conv_w_t <<<cg, 256, 0, stream>>>(W1, w1t, EDIM, FF);
    k_conv_w_t <<<cg, 256, 0, stream>>>(W2, w2t, FF, EDIM);
    k_conv_plain<<<cg, 256, 0, stream>>>(src, srcbf, (size_t)MROWS * EDIM);

    const dim3 gEE(EDIM / BN, MROWS / BM);   // (8, 64)
    const dim3 gEF(FF   / BN, MROWS / BM);   // (32, 64)
    const dim3 gfa(SEQ / 64, NH, BB);        // (16, 16, 8)

    // ---- self-attention block (pre-norm, causal) ----
    k_layernorm<<<MROWS, 256, 0, stream>>>(tgt, g1, be1, xn);
    k_gemm_bf16<0><<<gEE, 256, 0, stream>>>(xn, wq1, bq1, nullptr, qb, nullptr, MROWS, EDIM, EDIM);
    k_gemm_bf16<0><<<gEE, 256, 0, stream>>>(xn, wk1, bk1, nullptr, kb, nullptr, MROWS, EDIM, EDIM);
    k_gemm_bf16<0><<<gEE, 256, 0, stream>>>(xn, wv1, bv1, nullptr, vb, nullptr, MROWS, EDIM, EDIM);
    k_gemm_bf16<0><<<gEE, 256, 0, stream>>>(qb, whq1, bhq1, nullptr, qhb, nullptr, MROWS, EDIM, EDIM);
    k_gemm_bf16<0><<<gEE, 256, 0, stream>>>(kb, whk1, bhk1, nullptr, khb, nullptr, MROWS, EDIM, EDIM);
    k_gemm_bf16<0><<<gEE, 256, 0, stream>>>(vb, whv1, bhv1, nullptr, vhb, nullptr, MROWS, EDIM, EDIM);
    k_flash<true><<<gfa, 128, 0, stream>>>(qhb, khb, vhb, ob, SEQ, SEQ);
    k_gemm_bf16<2><<<gEE, 256, 0, stream>>>(ob, wo1, bo1, tgt, nullptr, r1, MROWS, EDIM, EDIM);

    // ---- cross-attention block (no mask) ----
    k_layernorm<<<MROWS, 256, 0, stream>>>(r1, g2, be2, xn);
    k_gemm_bf16<0><<<gEE, 256, 0, stream>>>(xn, wq2, bq2, nullptr, qb, nullptr, MROWS, EDIM, EDIM);
    k_gemm_bf16<0><<<gEE, 256, 0, stream>>>(qb, whq2, bhq2, nullptr, qhb, nullptr, MROWS, EDIM, EDIM);
    k_gemm_bf16<0><<<gEE, 256, 0, stream>>>(srcbf, wk2, bk2, nullptr, kb, nullptr, MROWS, EDIM, EDIM);
    k_gemm_bf16<0><<<gEE, 256, 0, stream>>>(kb, whk2, bhk2, nullptr, khb, nullptr, MROWS, EDIM, EDIM);
    k_gemm_bf16<0><<<gEE, 256, 0, stream>>>(srcbf, wv2, bv2, nullptr, vb, nullptr, MROWS, EDIM, EDIM);
    k_gemm_bf16<0><<<gEE, 256, 0, stream>>>(vb, whv2, bhv2, nullptr, vhb, nullptr, MROWS, EDIM, EDIM);
    k_flash<false><<<gfa, 128, 0, stream>>>(qhb, khb, vhb, ob, SEQ, SEQ);
    k_gemm_bf16<2><<<gEE, 256, 0, stream>>>(ob, wo2, bo2, r1, nullptr, r2, MROWS, EDIM, EDIM);

    // ---- FFN block (pre-norm, exact GELU) ----
    k_layernorm<<<MROWS, 256, 0, stream>>>(r2, g3, be3, xn);
    k_gemm_bf16<1><<<gEF, 256, 0, stream>>>(xn, w1t, b1, nullptr, hb, nullptr, MROWS, FF, EDIM);
    k_gemm_bf16<2><<<gEE, 256, 0, stream>>>(hb, w2t, b2, r2, nullptr, out, MROWS, EDIM, FF);
}
